// HMQSoftmax_59983513256165
// MI455X (gfx1250) — compile-verified, matrix operand order checked by reading, not run on hardware
//
#include <hip/hip_runtime.h>

typedef __attribute__((ext_vector_type(16))) __bf16          v16bf;
typedef __attribute__((ext_vector_type(8)))  float           v8f;
typedef __attribute__((ext_vector_type(16))) unsigned short  v16u;
typedef __attribute__((ext_vector_type(4)))  float           f4;

#define ROW_LEN          2048
#define THREADS          128     // 4 waves of 32 per row
#define ELEMS_PER_THREAD 16      // 128 * 16 = 2048
#define INV_LN2          1.4426950408889634f

// bf16 helpers (round-to-nearest-even) for the scalar isqrt path
__device__ __forceinline__ unsigned short f32_to_bf16(float f) {
    unsigned int u = __float_as_uint(f);
    u += 0x7FFFu + ((u >> 16) & 1u);
    return (unsigned short)(u >> 16);
}
__device__ __forceinline__ float bf16_to_f32(unsigned short h) {
    return __uint_as_float(((unsigned int)h) << 16);
}

__global__ void __launch_bounds__(THREADS)
hmq_softmax_kernel(const float* __restrict__ x, float* __restrict__ out) {
    const long long row = blockIdx.x;
    const float* xr   = x   + row * (long long)ROW_LEN;
    float*       outr = out + row * (long long)ROW_LEN;

    const int tid  = threadIdx.x;
    const int base = tid * ELEMS_PER_THREAD;

    // ---- phase 1: stream 16 f32/thread (NT: 1 GB stream >> 192 MB L2).
    // q = floor(x/ln2) is a small integer (|x| <~ 6.2 for the N(0,1) input),
    // so bf16(q) == q and exp2(q) is a power of two => bf16(exp2(q)) is just
    // the top 16 bits of the f32.  v_cvt_i32 + v_ldexp replaces v_exp + two
    // RTNE rounding sequences with bit-identical results on this domain.
    v16u ebits;
#pragma unroll
    for (int v = 0; v < 4; ++v) {
        f4 xv = __builtin_nontemporal_load((const f4*)(xr + base) + v);
#pragma unroll
        for (int j = 0; j < 4; ++j) {
            float q  = __builtin_floorf(xv[j] * INV_LN2);
            float ex = __builtin_amdgcn_ldexpf(1.0f, (int)q);  // exact 2^q
            ebits[v * 4 + j] = (unsigned short)(__float_as_uint(ex) >> 16);
        }
    }

    // ---- phase 2: row-chunk reduction on the matrix pipe.
    // This wave's 512 bf16 exp values are one 16x32 A-tile; B = all-ones, so
    // every column of D equals sum(A).
    v16bf A = __builtin_bit_cast(v16bf, ebits);
    v16u onebits;
#pragma unroll
    for (int i = 0; i < 16; ++i) onebits[i] = 0x3F80u;         // bf16 1.0
    v16bf B = __builtin_bit_cast(v16bf, onebits);

    v8f acc = {};
    // 8 args: (neg_a, A, neg_b, B, c_mod, C, reuse_a, reuse_b)
    acc = __builtin_amdgcn_wmma_f32_16x16x32_bf16(false, A, false, B,
                                                  (short)0, acc, false, false);
    float s = 0.0f;
#pragma unroll
    for (int i = 0; i < 8; ++i) s += acc[i];
    // C/D layout: lane L holds M=0..7 (L<16) / M=8..15 (L>=16) of column L%16.
    // One SWAPX16 swizzle (group-of-32, xor_mask=0x10, and_mask=0x1f) pairs the
    // two half-columns => full column sum == sum(A).  No /16 needed.
    s += __int_as_float(__builtin_amdgcn_ds_swizzle(__float_as_int(s), 0x401F));

    __shared__ float s_part[THREADS / 32];
    __shared__ float s_r;
    const int wave = tid >> 5;
    const int lane = tid & 31;
    if (lane == 0) s_part[wave] = s;
    __syncthreads();

    // ---- phase 3: bf16 Quake isqrt of the row sum (magic 24375 + 1 Newton step)
    if (tid == 0) {
        float row_sum = 0.0f;
#pragma unroll
        for (int w = 0; w < THREADS / 32; ++w) row_sum += s_part[w];
        unsigned short sb = f32_to_bf16(row_sum);
        short i16 = (short)sb;
        i16 = (short)(24375 - (i16 >> 1));                     // int16 shift + wrap
        float y   = bf16_to_f32((unsigned short)i16);          // initial estimate
        float xf  = bf16_to_f32(sb);
        float y2  = bf16_to_f32(f32_to_bf16(y * y));
        float xh  = bf16_to_f32(f32_to_bf16(xf * 0.5f));
        float mul = bf16_to_f32(f32_to_bf16(xh * y2));
        float sub = bf16_to_f32(f32_to_bf16(1.5f - mul));
        s_r = bf16_to_f32(f32_to_bf16(y * sub));               // one Newton iter
    }
    __syncthreads();
    const float r = s_r;

    // ---- phase 4: out = (exp_x * r) * r with bf16 rounding per step (native
    // __bf16 casts -> hardware bf16 converters), stored as f32 NT.
#pragma unroll
    for (int v = 0; v < 4; ++v) {
        f4 o;
#pragma unroll
        for (int j = 0; j < 4; ++j) {
            float  e = bf16_to_f32(ebits[v * 4 + j]);
            __bf16 t = (__bf16)(e * r);
            __bf16 u = (__bf16)((float)t * r);
            o[j] = (float)u;
        }
        __builtin_nontemporal_store(o, (f4*)(outr + base) + v);
    }
}

extern "C" void kernel_launch(void* const* d_in, const int* in_sizes, int n_in,
                              void* d_out, int out_size, void* d_ws, size_t ws_size,
                              hipStream_t stream) {
    (void)n_in; (void)d_ws; (void)ws_size; (void)out_size;
    const float* x   = (const float*)d_in[0];
    float*       out = (float*)d_out;
    const int n    = in_sizes[0];       // 2*16*2048*2048
    const int rows = n / ROW_LEN;       // 65536 rows
    hmq_softmax_kernel<<<rows, THREADS, 0, stream>>>(x, out);
}
